// GNN_node_Virtualnode_64183991272049
// MI455X (gfx1250) — matrix operand order repaired and checked by reading, not compiled, not run on hardware
//
#include <hip/hip_runtime.h>
#include <hip/hip_bf16.h>

#define NN 100000
#define EE 600000
#define DD 128
#define D2 256
#define LL 5
#define GG 512
#define BN_EPS 1e-5f

typedef _Float16 v16h __attribute__((ext_vector_type(16)));
typedef _Float16 v8h  __attribute__((ext_vector_type(8)));
typedef float    v8f  __attribute__((ext_vector_type(8)));

// ---------------- elementwise / gather / scatter kernels ----------------

__global__ void embed_kernel(const int* __restrict__ x, const float* __restrict__ emb,
                             float* __restrict__ h, int total) {
    int i = blockIdx.x * blockDim.x + threadIdx.x;
    if (i >= total) return;
    int node = i >> 7, d = i & 127;
    h[i] = emb[x[node] * DD + d];
}

__global__ void vn_init_kernel(const float* __restrict__ vn_emb, float* __restrict__ vn, int total) {
    int i = blockIdx.x * blockDim.x + threadIdx.x;
    if (i >= total) return;
    vn[i] = vn_emb[i & 127];
}

// h_in = h + vn[batch];  z = (1+eps[l]) * h_in
__global__ void hin_z_kernel(const float* __restrict__ h, const float* __restrict__ vn,
                             const int* __restrict__ batch, const float* __restrict__ eps,
                             int layer, float* __restrict__ h_in, float* __restrict__ z, int total) {
    int i = blockIdx.x * blockDim.x + threadIdx.x;
    if (i >= total) return;
    int node = i >> 7, d = i & 127;
    float v = h[i] + vn[batch[node] * DD + d];
    h_in[i] = v;
    z[i] = (1.0f + eps[layer]) * v;
}

// z[dst[e]] += h_in[src[e]]   (per-edge, per-feature atomics)
__global__ void edge_scatter_kernel(const int* __restrict__ src, const int* __restrict__ dst,
                                    const float* __restrict__ h_in, float* __restrict__ z, int total) {
    int i = blockIdx.x * blockDim.x + threadIdx.x;
    if (i >= total) return;
    int e = i >> 7, d = i & 127;
    atomicAdd(&z[(size_t)dst[e] * DD + d], h_in[(size_t)src[e] * DD + d]);
}

__global__ void copy_kernel(const float* __restrict__ in, float* __restrict__ out, int total) {
    int i = blockIdx.x * blockDim.x + threadIdx.x;
    if (i >= total) return;
    out[i] = in[i];
}

// vtemp[batch[node]] += h_in[node]  (vtemp pre-initialized with vn)
__global__ void segsum_kernel(const int* __restrict__ batch, const float* __restrict__ h_in,
                              float* __restrict__ vtemp, int total) {
    int i = blockIdx.x * blockDim.x + threadIdx.x;
    if (i >= total) return;
    int node = i >> 7, d = i & 127;
    atomicAdd(&vtemp[batch[node] * DD + d], h_in[i]);
}

__global__ void cvt_f32_to_f16_kernel(const float* __restrict__ in, _Float16* __restrict__ out, int total) {
    int i = blockIdx.x * blockDim.x + threadIdx.x;
    if (i >= total) return;
    out[i] = (_Float16)in[i];
}

// ---------------- weight packing into WMMA B-fragment layout ----------------
// Packed layout: for col-tile nt, K-chunk kc, lane L (0..31), 16 halves:
//   b[s] = W[K = kc*32 + (s<8 ? (L>>4)*8 + s : 16 + (L>>4)*8 + s-8), N = nt*16 + (L&15)]
__global__ void pack_b_kernel(const float* __restrict__ W, _Float16* __restrict__ Wp,
                              int K, int Nout) {
    int idx = blockIdx.x * blockDim.x + threadIdx.x;
    if (idx >= K * Nout) return;
    int k = idx / Nout, n = idx % Nout;
    int kc = k >> 5, kk = k & 31;
    int half = (kk >> 3) & 1, r = (kk & 7) | ((kk >> 4) << 3);
    int v = r >> 1, hb = r & 1;
    int lane = half * 16 + (n & 15);
    int nt = n >> 4;
    int KC = K >> 5;
    size_t dstIdx = ((size_t)(nt * KC + kc) * 32 + lane) * 16 + v * 2 + hb;
    Wp[dstIdx] = (_Float16)W[idx];
}

// ---------------- WMMA GEMM + fused bias/BN/ReLU epilogue ----------------
// C[M,Nout] = A[M,KC*32] (row-major f16) @ Bpacked ; per-col bias + eval-BN
// + optional ReLU; output f32 (outF) or f16 (outH).
// Software-pipelined: loads for chunk k+1 issue before the WMMA of chunk k.
template <int KC>
__global__ __launch_bounds__(256)
void gemm_wmma_kernel(const _Float16* __restrict__ A, const _Float16* __restrict__ Bp,
                      const float* __restrict__ bias,
                      const float* __restrict__ bng, const float* __restrict__ bnb,
                      const float* __restrict__ bnm, const float* __restrict__ bnv,
                      float* __restrict__ outF, _Float16* __restrict__ outH,
                      int Nout, int relu) {
    const int K = KC * 32;
    const int lane = threadIdx.x & 31;
    const int wave = threadIdx.x >> 5;
    const int rowTile = blockIdx.x;
    const int colTile = blockIdx.y * 8 + wave;
    const int nColTiles = Nout >> 4;
    if (colTile >= nColTiles) return;

    const int m0 = rowTile << 4;
    const int half = lane >> 4;
    const int nloc = lane & 15;
    const int n = (colTile << 4) + nloc;
    const int arow = m0 + nloc;            // A-fragment: M = lane%16 for both halves

    const _Float16* Abase = A + (size_t)arow * K + half * 8;
    const _Float16* Bbase = Bp + ((size_t)colTile * KC * 32 + lane) * 16;

    auto loadA = [&](int kc) -> v16h {
        v8h lo = *(const v8h*)(Abase + kc * 32);        // K base .. base+7
        v8h hi = *(const v8h*)(Abase + kc * 32 + 16);   // K base+16 .. base+23
        return __builtin_shufflevector(lo, hi, 0, 1, 2, 3, 4, 5, 6, 7,
                                       8, 9, 10, 11, 12, 13, 14, 15);
    };
    auto loadB = [&](int kc) -> v16h {
        return *(const v16h*)(Bbase + (size_t)kc * 32 * 16);
    };

    v16h abuf[2], bbuf[2];
    abuf[0] = loadA(0);
    bbuf[0] = loadB(0);

    v8f acc = {};
    #pragma unroll
    for (int kc = 0; kc < KC; ++kc) {
        const int cur = kc & 1;
        const int nxt = cur ^ 1;
        if (kc + 2 < KC) {  // prefetch two chunks ahead
            __builtin_prefetch((const void*)(Abase + (kc + 2) * 32), 0, 3);
            __builtin_prefetch((const void*)(Bbase + (size_t)(kc + 2) * 32 * 16), 0, 3);
        }
        if (kc + 1 < KC) {  // double-buffered loads in flight over the WMMA
            abuf[nxt] = loadA(kc + 1);
            bbuf[nxt] = loadB(kc + 1);
        }
        acc = __builtin_amdgcn_wmma_f32_16x16x32_f16(
            false, abuf[cur], false, bbuf[cur], (short)0, acc, false, false);
    }

    // per-column epilogue params
    float bi = bias[n];
    float g = bng[n], be = bnb[n], mu = bnm[n];
    float inv = rsqrtf(bnv[n] + BN_EPS);

    #pragma unroll
    for (int r = 0; r < 8; ++r) {
        int m = m0 + r + 8 * half;       // C/D layout: lanes16-31 hold M = r+8
        float v = acc[r] + bi;
        v = (v - mu) * inv * g + be;
        if (relu) v = v > 0.0f ? v : 0.0f;
        size_t o = (size_t)m * Nout + n;
        if (outF) outF[o] = v;
        else      outH[o] = (_Float16)v;
    }
}

// ---------------- launch ----------------

static inline size_t align256(size_t x) { return (x + 255) & ~(size_t)255; }

extern "C" void kernel_launch(void* const* d_in, const int* in_sizes, int n_in,
                              void* d_out, int out_size, void* d_ws, size_t ws_size,
                              hipStream_t stream) {
    (void)in_sizes; (void)n_in; (void)out_size; (void)ws_size;

    const int*   x       = (const int*)d_in[0];
    const int*   ei      = (const int*)d_in[1];
    const int*   batch   = (const int*)d_in[2];
    const float* enc_emb = (const float*)d_in[3];
    const float* vn_emb  = (const float*)d_in[4];
    const float* eps     = (const float*)d_in[5];
    const float* mW1     = (const float*)d_in[6];
    const float* mb1     = (const float*)d_in[7];
    const float* mbn_g   = (const float*)d_in[8];
    const float* mbn_b   = (const float*)d_in[9];
    const float* mbn_m   = (const float*)d_in[10];
    const float* mbn_v   = (const float*)d_in[11];
    const float* mW2     = (const float*)d_in[12];
    const float* mb2     = (const float*)d_in[13];
    const float* bn_g    = (const float*)d_in[14];
    const float* bn_b    = (const float*)d_in[15];
    const float* bn_m    = (const float*)d_in[16];
    const float* bn_v    = (const float*)d_in[17];
    const float* vW1     = (const float*)d_in[18];
    const float* vb1     = (const float*)d_in[19];
    const float* vbn1_g  = (const float*)d_in[20];
    const float* vbn1_b  = (const float*)d_in[21];
    const float* vbn1_m  = (const float*)d_in[22];
    const float* vbn1_v  = (const float*)d_in[23];
    const float* vW2     = (const float*)d_in[24];
    const float* vb2     = (const float*)d_in[25];
    const float* vbn2_g  = (const float*)d_in[26];
    const float* vbn2_b  = (const float*)d_in[27];
    const float* vbn2_m  = (const float*)d_in[28];
    const float* vbn2_v  = (const float*)d_in[29];

    const int* esrc = ei;
    const int* edst = ei + EE;

    // workspace carve-up
    char* base = (char*)d_ws;
    size_t off = 0;
    float*    h     = (float*)(base + off);    off += align256((size_t)NN * DD * 4);
    float*    h_in  = (float*)(base + off);    off += align256((size_t)NN * DD * 4);
    float*    z     = (float*)(base + off);    off += align256((size_t)NN * DD * 4);
    _Float16* z16   = (_Float16*)(base + off); off += align256((size_t)NN * DD * 2);
    _Float16* Y1    = (_Float16*)(base + off); off += align256((size_t)NN * D2 * 2);
    float*    vn    = (float*)(base + off);    off += align256((size_t)GG * DD * 4);
    float*    vtemp = (float*)(base + off);    off += align256((size_t)GG * DD * 4);
    _Float16* vt16  = (_Float16*)(base + off); off += align256((size_t)GG * DD * 2);
    _Float16* Yv    = (_Float16*)(base + off); off += align256((size_t)GG * D2 * 2);
    _Float16* W1p   = (_Float16*)(base + off); off += align256((size_t)LL * DD * D2 * 2);
    _Float16* W2p   = (_Float16*)(base + off); off += align256((size_t)LL * D2 * DD * 2);
    _Float16* vW1p  = (_Float16*)(base + off); off += align256((size_t)(LL - 1) * DD * D2 * 2);
    _Float16* vW2p  = (_Float16*)(base + off); off += align256((size_t)(LL - 1) * D2 * DD * 2);

    const int T = 256;
    const int ND  = NN * DD;       // 12.8M
    const int ED  = EE * DD;       // 76.8M
    const int GD  = GG * DD;
    const int WKN = DD * D2;       // 32768 elements per weight matrix

    // pack all weights into WMMA B-fragment layout (deterministic each call)
    for (int l = 0; l < LL; ++l) {
        pack_b_kernel<<<(WKN + T - 1) / T, T, 0, stream>>>(mW1 + (size_t)l * WKN, W1p + (size_t)l * WKN, DD, D2);
        pack_b_kernel<<<(WKN + T - 1) / T, T, 0, stream>>>(mW2 + (size_t)l * WKN, W2p + (size_t)l * WKN, D2, DD);
    }
    for (int l = 0; l < LL - 1; ++l) {
        pack_b_kernel<<<(WKN + T - 1) / T, T, 0, stream>>>(vW1 + (size_t)l * WKN, vW1p + (size_t)l * WKN, DD, D2);
        pack_b_kernel<<<(WKN + T - 1) / T, T, 0, stream>>>(vW2 + (size_t)l * WKN, vW2p + (size_t)l * WKN, D2, DD);
    }

    // encoders
    embed_kernel<<<(ND + T - 1) / T, T, 0, stream>>>(x, enc_emb, h, ND);
    vn_init_kernel<<<(GD + T - 1) / T, T, 0, stream>>>(vn_emb, vn, GD);

    for (int l = 0; l < LL; ++l) {
        // h_in = h + vn[batch]; z = (1+eps)*h_in
        hin_z_kernel<<<(ND + T - 1) / T, T, 0, stream>>>(h, vn, batch, eps, l, h_in, z, ND);
        // z[dst] += h_in[src]
        edge_scatter_kernel<<<(ED + T - 1) / T, T, 0, stream>>>(esrc, edst, h_in, z, ED);

        if (l < LL - 1) {
            // vtemp = segment_sum(h_in, batch) + vn   (init with vn, atomic add)
            copy_kernel<<<(GD + T - 1) / T, T, 0, stream>>>(vn, vtemp, GD);
            segsum_kernel<<<(ND + T - 1) / T, T, 0, stream>>>(batch, h_in, vtemp, ND);
        }

        // GIN MLP on nodes
        cvt_f32_to_f16_kernel<<<(ND + T - 1) / T, T, 0, stream>>>(z, z16, ND);
        // Y1 = relu(BN(z @ W1 + b1)) -> f16 [N, 256]
        gemm_wmma_kernel<4><<<dim3(NN / 16, 2), 256, 0, stream>>>(
            z16, W1p + (size_t)l * WKN, mb1 + (size_t)l * D2,
            mbn_g + (size_t)l * D2, mbn_b + (size_t)l * D2,
            mbn_m + (size_t)l * D2, mbn_v + (size_t)l * D2,
            nullptr, Y1, D2, 1);
        // h = BN(Y1 @ W2 + b2); relu except last layer; f32 -> h (or d_out)
        float* outH = (l == LL - 1) ? (float*)d_out : h;
        gemm_wmma_kernel<8><<<dim3(NN / 16, 1), 256, 0, stream>>>(
            Y1, W2p + (size_t)l * WKN, mb2 + (size_t)l * DD,
            bn_g + (size_t)l * DD, bn_b + (size_t)l * DD,
            bn_m + (size_t)l * DD, bn_v + (size_t)l * DD,
            outH, nullptr, DD, (l < LL - 1) ? 1 : 0);

        if (l < LL - 1) {
            // virtual-node MLP: vn = relu(BN2( relu(BN1(vtemp@vW1+vb1)) @ vW2 + vb2 ))
            cvt_f32_to_f16_kernel<<<(GD + T - 1) / T, T, 0, stream>>>(vtemp, vt16, GD);
            gemm_wmma_kernel<4><<<dim3(GG / 16, 2), 256, 0, stream>>>(
                vt16, vW1p + (size_t)l * WKN, vb1 + (size_t)l * D2,
                vbn1_g + (size_t)l * D2, vbn1_b + (size_t)l * D2,
                vbn1_m + (size_t)l * D2, vbn1_v + (size_t)l * D2,
                nullptr, Yv, D2, 1);
            gemm_wmma_kernel<8><<<dim3(GG / 16, 1), 256, 0, stream>>>(
                Yv, vW2p + (size_t)l * WKN, vb2 + (size_t)l * DD,
                vbn2_g + (size_t)l * DD, vbn2_b + (size_t)l * DD,
                vbn2_m + (size_t)l * DD, vbn2_v + (size_t)l * DD,
                vn, nullptr, DD, 1);
        }
    }
}
